// SAPblock_74028056314283
// MI455X (gfx1250) — compile-verified, matrix operand order checked by reading, not run on hardware
//
#include <hip/hip_runtime.h>
#include <hip/hip_bf16.h>

// ---------------------------------------------------------------------------
// Problem constants (reference: B=2, C=512, H=W=96)
// ---------------------------------------------------------------------------
#define HWC 9216            // H*W
#define NPOS 18432          // B*H*W

typedef __attribute__((ext_vector_type(16))) __bf16 v16bf;
typedef __attribute__((ext_vector_type(8)))  float  v8f;

__device__ __forceinline__ unsigned short f32_to_bf16(float f) {
  unsigned int u = __float_as_uint(f);
  unsigned int r = u + 0x7FFFu + ((u >> 16) & 1u);   // round-to-nearest-even
  return (unsigned short)(r >> 16);
}
__device__ __forceinline__ __bf16 us_to_bf16(unsigned short s) {
  union { unsigned short u; __bf16 b; } cv; cv.u = s; return cv.b;
}
__device__ __forceinline__ v8f wmma_bf16(v16bf a, v16bf b, v8f c) {
  return __builtin_amdgcn_wmma_f32_16x16x32_bf16(false, a, false, b, (short)0, c,
                                                 false, false);
}

// ---------------------------------------------------------------------------
// f32 -> bf16 weight conversion
// ---------------------------------------------------------------------------
__global__ void cvt_bf16_kernel(const float* __restrict__ in,
                                unsigned short* __restrict__ out, int n) {
  int i = blockIdx.x * 256 + threadIdx.x;
  if (i < n) out[i] = f32_to_bf16(in[i]);
}

// ---------------------------------------------------------------------------
// Shared WMMA tile core:
//   WG = 128 threads (4 waves). Tile = 64(M) x 256(N), K-step 32.
//   Each wave owns a 64x64 sub-tile: 4x4 fragments, 16 WMMA per k-step.
//   As[m][k]  : 64x32 bf16 (weights), rows contiguous -> ds_load_b128 frags
//   BsT[n][k] : 256x32 bf16 (acts, transposed) -> ds_load_b128 frags too
// ---------------------------------------------------------------------------
#define GEMM_DECLS                                                     \
  __shared__ __align__(16) unsigned short As[64][32];                  \
  __shared__ __align__(16) unsigned short BsT[256][32];                \
  const int tid  = threadIdx.x;                                        \
  const int lane = tid & 31, wave = tid >> 5;                          \
  const int m0 = blockIdx.x * 64, n0 = blockIdx.y * 256;               \
  const int lm  = lane & 15;                                           \
  const int khi = (lane & 16) >> 1; /* ISA 16-bit frag K offset */     \
  const int hi  = lane >> 4;                                           \
  const int nw  = wave * 64;                                           \
  v8f zero = {0.f, 0.f, 0.f, 0.f, 0.f, 0.f, 0.f, 0.f};                 \
  v8f acc[4][4];                                                       \
  _Pragma("unroll") for (int i = 0; i < 4; ++i)                        \
  _Pragma("unroll") for (int j = 0; j < 4; ++j) acc[i][j] = zero;

#define GEMM_COMPUTE                                                   \
  {                                                                    \
    v16bf afr[4];                                                      \
    _Pragma("unroll") for (int f = 0; f < 4; ++f)                      \
      _Pragma("unroll") for (int h = 0; h < 16; ++h) {                 \
        int kk = h + (h & 8) + khi;                                    \
        afr[f][h] = us_to_bf16(As[f * 16 + lm][kk]);                   \
      }                                                                \
    _Pragma("unroll") for (int fn = 0; fn < 4; ++fn) {                 \
      v16bf bfr;                                                       \
      _Pragma("unroll") for (int h = 0; h < 16; ++h) {                 \
        int kk = h + (h & 8) + khi;                                    \
        bfr[h] = us_to_bf16(BsT[nw + fn * 16 + lm][kk]);               \
      }                                                                \
      acc[0][fn] = wmma_bf16(afr[0], bfr, acc[0][fn]);                 \
      acc[1][fn] = wmma_bf16(afr[1], bfr, acc[1][fn]);                 \
      acc[2][fn] = wmma_bf16(afr[2], bfr, acc[2][fn]);                 \
      acc[3][fn] = wmma_bf16(afr[3], bfr, acc[3][fn]);                 \
    }                                                                  \
  }

#define GEMM_EPILOGUE(MM)                                              \
  _Pragma("unroll") for (int fm = 0; fm < 4; ++fm)                     \
  _Pragma("unroll") for (int fn = 0; fn < 4; ++fn)                     \
  _Pragma("unroll") for (int r = 0; r < 8; ++r) {                      \
    int m = m0 + fm * 16 + r + 8 * hi;                                 \
    int n = n0 + nw + fn * 16 + lm;                                    \
    float v = acc[fm][fn][r];                                          \
    if (bias) v += bias[m];                                            \
    if (relu) v = fmaxf(v, 0.f);                                       \
    int b = (n >= HWC) ? 1 : 0;                                        \
    int hw = n - b * HWC;                                              \
    out[((size_t)(b * (MM) + m)) * HWC + hw] = v;                      \
  }

// ---------------------------------------------------------------------------
// WMMA GEMM for 1x1 convs:  out[b,m,hw] = sum_k W[m,k] * act[b,k,hw]
// act channel k comes from src0 (k<K0) or src1 (k-K0); concat along channels.
// ---------------------------------------------------------------------------
__global__ __launch_bounds__(128) void gemm1x1_kernel(
    const unsigned short* __restrict__ Wbf, const float* __restrict__ src0,
    const float* __restrict__ src1, const float* __restrict__ bias,
    float* __restrict__ out, int M, int K, int K0, int relu) {
  GEMM_DECLS
  const int K1 = K - K0;

  for (int k0t = 0; k0t < K; k0t += 32) {
#pragma unroll
    for (int i = 0; i < 16; ++i) {            // A tile: 64x32 bf16 weights
      int idx = i * 128 + tid;
      int mi = idx >> 5, kk = idx & 31;
      As[mi][kk] = Wbf[(size_t)(m0 + mi) * K + (k0t + kk)];
    }
#pragma unroll 4
    for (int i = 0; i < 64; ++i) {            // B tile: 32k x 256n, f32->bf16
      int idx = i * 128 + tid;
      int kk = idx >> 8, nn = idx & 255;
      int n = n0 + nn;
      int b = (n >= HWC) ? 1 : 0;
      int hw = n - b * HWC;
      int k = k0t + kk;
      float v;
      if (k < K0) v = src0[((size_t)(b * K0 + k)) * HWC + hw];
      else        v = src1[((size_t)(b * K1 + (k - K0))) * HWC + hw];
      BsT[nn][kk] = f32_to_bf16(v);
    }
    __syncthreads();
    GEMM_COMPUTE
    __syncthreads();
  }
  GEMM_EPILOGUE(M)
}

// ---------------------------------------------------------------------------
// WMMA implicit-GEMM 3x3 conv (dilated, padded): K = Cin*9, im2col on the fly
// ---------------------------------------------------------------------------
__global__ __launch_bounds__(128) void gemm3x3_kernel(
    const unsigned short* __restrict__ Wbf, const float* __restrict__ src,
    const float* __restrict__ bias, float* __restrict__ out,
    int M, int Cin, int dil, int relu) {
  GEMM_DECLS
  const int K = Cin * 9;

  for (int k0t = 0; k0t < K; k0t += 32) {
#pragma unroll
    for (int i = 0; i < 16; ++i) {
      int idx = i * 128 + tid;
      int mi = idx >> 5, kk = idx & 31;
      As[mi][kk] = Wbf[(size_t)(m0 + mi) * K + (k0t + kk)];
    }
#pragma unroll 4
    for (int i = 0; i < 64; ++i) {
      int idx = i * 128 + tid;
      int kk = idx >> 8, nn = idx & 255;
      int k = k0t + kk;
      int ci = k / 9;
      int rs = k - ci * 9;
      int rr = rs / 3;
      int dh = (rr - 1) * dil;
      int dw = (rs - rr * 3 - 1) * dil;
      int n = n0 + nn;
      int b = (n >= HWC) ? 1 : 0;
      int hw = n - b * HWC;
      int h = hw / 96;
      int w = hw - h * 96;
      int ih = h + dh, iw = w + dw;
      float v = 0.f;
      if ((unsigned)ih < 96u && (unsigned)iw < 96u)
        v = src[((size_t)(b * Cin + ci)) * HWC + ih * 96 + iw];
      BsT[nn][kk] = f32_to_bf16(v);
    }
    __syncthreads();
    GEMM_COMPUTE
    __syncthreads();
  }
  GEMM_EPILOGUE(M)
}

// ---------------------------------------------------------------------------
// BatchNorm: stats (mean, 1/sqrt(var+eps)) per channel, then apply
// ---------------------------------------------------------------------------
__global__ __launch_bounds__(256) void bn_stats_kernel(
    const float* __restrict__ x, float* __restrict__ stats, int C) {
  __shared__ float s1[256], s2[256];
  int c = blockIdx.x, tid = threadIdx.x;
  float a = 0.f, b2 = 0.f;
  for (int bb = 0; bb < 2; ++bb) {
    const float* p = x + ((size_t)(bb * C + c)) * HWC;
    for (int i = tid; i < HWC; i += 256) { float v = p[i]; a += v; b2 += v * v; }
  }
  s1[tid] = a; s2[tid] = b2; __syncthreads();
  for (int s = 128; s > 0; s >>= 1) {
    if (tid < s) { s1[tid] += s1[tid + s]; s2[tid] += s2[tid + s]; }
    __syncthreads();
  }
  if (tid == 0) {
    float m = s1[0] / (float)NPOS;
    float var = s2[0] / (float)NPOS - m * m;
    stats[2 * c] = m;
    stats[2 * c + 1] = rsqrtf(var + 1e-5f);
  }
}

__global__ void bn_apply_kernel(const float* __restrict__ x,
                                const float* __restrict__ stats,
                                const float* __restrict__ g,
                                const float* __restrict__ bt,
                                float* __restrict__ out, int C, int relu) {
  size_t i = (size_t)blockIdx.x * 256 + threadIdx.x;
  if (i >= (size_t)2 * C * HWC) return;
  int c = (int)((i / HWC) % C);
  float v = (x[i] - stats[2 * c]) * stats[2 * c + 1] * g[c] + bt[c];
  if (relu) v = fmaxf(v, 0.f);
  out[i] = v;
}

// ---------------------------------------------------------------------------
// c32 3x3 conv (256->2, pad=1) + channel softmax -> att[b,2,hw]
// ---------------------------------------------------------------------------
__global__ __launch_bounds__(256) void c32_softmax_kernel(
    const float* __restrict__ feat, const float* __restrict__ wgt,
    const float* __restrict__ bias, float* __restrict__ att) {
  __shared__ float ws[4608];
  int tid = threadIdx.x;
  for (int i = tid; i < 4608; i += 256) ws[i] = wgt[i];
  __syncthreads();
  int n = blockIdx.x * 256 + tid;
  int b = (n >= HWC) ? 1 : 0;
  int hw = n - b * HWC;
  int h = hw / 96, w = hw - (hw / 96) * 96;
  float e0 = bias[0], e1 = bias[1];
  for (int ci = 0; ci < 256; ++ci) {
    const float* fp = feat + ((size_t)(b * 256 + ci)) * HWC;
    const float* w0 = ws + ci * 9;
#pragma unroll
    for (int rs = 0; rs < 9; ++rs) {
      int ih = h + rs / 3 - 1, iw = w + rs % 3 - 1;
      if ((unsigned)ih < 96u && (unsigned)iw < 96u) {
        float f = fp[ih * 96 + iw];
        e0 += f * w0[rs];
        e1 += f * w0[2304 + rs];
      }
    }
  }
  float m = fmaxf(e0, e1);
  float x0 = __expf(e0 - m), x1 = __expf(e1 - m);
  float s = x0 + x1;
  att[((size_t)(b * 2)) * HWC + hw] = x0 / s;
  att[((size_t)(b * 2 + 1)) * HWC + hw] = x1 / s;
}

// ---------------------------------------------------------------------------
// Criss-cross: per-pixel energies over H column + W row, softmax over 192
// ---------------------------------------------------------------------------
__global__ __launch_bounds__(192) void cc_attn_kernel(
    const float* __restrict__ q, const float* __restrict__ k,
    float* __restrict__ a) {
  __shared__ float qv[64];
  __shared__ float red[256];
  int pos = blockIdx.x;
  int b = pos / HWC;
  int hw = pos - b * HWC;
  int h = hw / 96, w = hw - h * 96;
  int tid = threadIdx.x;
  if (tid < 64) qv[tid] = q[((size_t)(b * 64 + tid)) * HWC + hw];
  __syncthreads();
  float e;
  if (tid < 96) {
    int p = tid;
    if (p == h) e = -3.0e38f;                 // diagonal mask (-inf)
    else {
      e = 0.f;
      for (int c = 0; c < 64; ++c)
        e += qv[c] * k[((size_t)(b * 64 + c)) * HWC + p * 96 + w];
    }
  } else {
    int qq = tid - 96;
    e = 0.f;
    for (int c = 0; c < 64; ++c)
      e += qv[c] * k[((size_t)(b * 64 + c)) * HWC + h * 96 + qq];
  }
  red[tid] = e;
  if (tid < 64) red[192 + tid] = -3.0e38f;
  __syncthreads();
  for (int s = 128; s > 0; s >>= 1) {
    if (tid < s) red[tid] = fmaxf(red[tid], red[tid + s]);
    __syncthreads();
  }
  float m = red[0];
  __syncthreads();
  float ex = __expf(e - m);
  red[tid] = ex;
  if (tid < 64) red[192 + tid] = 0.f;
  __syncthreads();
  for (int s = 128; s > 0; s >>= 1) {
    if (tid < s) red[tid] += red[tid + s];
    __syncthreads();
  }
  a[(size_t)pos * 192 + tid] = ex / red[0];
}

__global__ __launch_bounds__(256) void cc_out_kernel(
    const float* __restrict__ v, const float* __restrict__ a,
    const float* __restrict__ res, const float* __restrict__ gamma,
    float* __restrict__ out, int relu) {
  size_t i = (size_t)blockIdx.x * 256 + threadIdx.x;
  if (i >= (size_t)2 * 512 * HWC) return;
  int b = (int)(i / ((size_t)512 * HWC));
  int rem = (int)(i - (size_t)b * 512 * HWC);
  int c = rem / HWC;
  int hw = rem - c * HWC;
  int h = hw / 96, w = hw - h * 96;
  const float* ap = a + ((size_t)(b * HWC + hw)) * 192;
  const float* vb = v + ((size_t)(b * 512 + c)) * HWC;
  float oh = 0.f, ow = 0.f;
  for (int p = 0; p < 96; ++p) oh += vb[p * 96 + w] * ap[p];
  for (int qq = 0; qq < 96; ++qq) ow += vb[h * 96 + qq] * ap[96 + qq];
  float r = gamma[0] * (oh + ow) + res[i];
  if (relu) r = fmaxf(r, 0.f);
  out[i] = r;
}

// ---------------------------------------------------------------------------
// Elementwise fusions
// ---------------------------------------------------------------------------
__global__ void mul_att_kernel(const float* __restrict__ att, int ch,
                               const float* __restrict__ src,
                               float* __restrict__ out) {
  size_t i = (size_t)blockIdx.x * 256 + threadIdx.x;
  if (i >= (size_t)2 * 512 * HWC) return;
  int b = (int)(i / ((size_t)512 * HWC));
  int hw = (int)(i % HWC);
  out[i] = src[i] * att[((size_t)(b * 2 + ch)) * HWC + hw];
}

__global__ void fuse_sap_kernel(const float* __restrict__ att,
                                const float* __restrict__ fus,
                                const float* __restrict__ b3,
                                const float* __restrict__ x,
                                const float* __restrict__ sg,
                                float* __restrict__ t, float* __restrict__ u) {
  size_t i = (size_t)blockIdx.x * 256 + threadIdx.x;
  if (i >= (size_t)2 * 512 * HWC) return;
  int b = (int)(i / ((size_t)512 * HWC));
  int hw = (int)(i % HWC);
  float g = sg[0];
  float a0 = att[((size_t)(b * 2)) * HWC + hw];
  float a1 = att[((size_t)(b * 2 + 1)) * HWC + hw];
  t[i] = g * (a0 * fus[i] + a1 * b3[i]);
  u[i] = (1.f - g) * x[i];
}

// ---------------------------------------------------------------------------
// Launcher
// ---------------------------------------------------------------------------
extern "C" void kernel_launch(void* const* d_in, const int* in_sizes, int n_in,
                              void* d_out, int out_size, void* d_ws,
                              size_t ws_size, hipStream_t stream) {
  (void)in_sizes; (void)n_in; (void)out_size; (void)ws_size;
  const float* x       = (const float*)d_in[0];
  const float* w3      = (const float*)d_in[1];
  const float* b3x3    = (const float*)d_in[2];
  const float* bn0g    = (const float*)d_in[3];
  const float* bn0b    = (const float*)d_in[4];
  const float* bn1g    = (const float*)d_in[5];
  const float* bn1b    = (const float*)d_in[6];
  const float* bn2g    = (const float*)d_in[7];
  const float* bn2b    = (const float*)d_in[8];
  const float* w1      = (const float*)d_in[9];
  const float* b1x1    = (const float*)d_in[10];
  const float* c31w    = (const float*)d_in[11];
  const float* c31b    = (const float*)d_in[12];
  const float* c32w    = (const float*)d_in[13];
  const float* c32b    = (const float*)d_in[14];
  const float* lastw   = (const float*)d_in[15];
  const float* lastg   = (const float*)d_in[16];
  const float* lastb   = (const float*)d_in[17];
  const float* sapg    = (const float*)d_in[18];
  const float* qw      = (const float*)d_in[19];
  const float* qb      = (const float*)d_in[20];
  const float* kw      = (const float*)d_in[21];
  const float* kb      = (const float*)d_in[22];
  const float* vw      = (const float*)d_in[23];
  const float* vb      = (const float*)d_in[24];
  const float* ccg     = (const float*)d_in[25];

  char* ws = (char*)d_ws;
  size_t off = 0;
  auto alloc = [&](size_t bytes) -> void* {
    void* p = ws + off;
    off += (bytes + 255) & ~(size_t)255;
    return p;
  };
  const size_t T = (size_t)2 * 512 * HWC;      // elements of a [2,512,96,96]

  unsigned short* W3  = (unsigned short*)alloc(512 * 512 * 9 * 2);
  unsigned short* W1  = (unsigned short*)alloc(512 * 1024 * 2);
  unsigned short* W31 = (unsigned short*)alloc(256 * 512 * 9 * 2);
  unsigned short* WL  = (unsigned short*)alloc(512 * 512 * 2);
  unsigned short* WQ  = (unsigned short*)alloc(64 * 512 * 2);
  unsigned short* WK  = (unsigned short*)alloc(64 * 512 * 2);
  unsigned short* WV  = (unsigned short*)alloc(512 * 512 * 2);
  float* STATS = (float*)alloc(512 * 2 * 4);
  float* B1f   = (float*)alloc(T * 4);
  float* B2f   = (float*)alloc(T * 4);
  float* B3f   = (float*)alloc(T * 4);
  float* F512  = (float*)alloc(T * 4);
  float* F256  = (float*)alloc(T * 2);
  float* ATT   = (float*)alloc((size_t)2 * 2 * HWC * 4);
  float* XCC   = (float*)alloc(T * 4);
  float* YCC   = (float*)alloc(T * 4);
  float* Qb    = (float*)alloc((size_t)2 * 64 * HWC * 4);
  float* Kb    = (float*)alloc((size_t)2 * 64 * HWC * 4);
  float* Vf    = (float*)alloc(T * 4);
  float* A192  = (float*)alloc((size_t)NPOS * 192 * 4);
  float* FUS   = (float*)alloc(T * 4);

  auto cvt = [&](const float* s, unsigned short* d, int n) {
    cvt_bf16_kernel<<<(n + 255) / 256, 256, 0, stream>>>(s, d, n);
  };
  cvt(w3, W3, 512 * 512 * 9);
  cvt(w1, W1, 512 * 1024);
  cvt(c31w, W31, 256 * 512 * 9);
  cvt(lastw, WL, 512 * 512);
  cvt(qw, WQ, 64 * 512);
  cvt(kw, WK, 64 * 512);
  cvt(vw, WV, 512 * 512);

  const dim3 g512(8, 72), g256(4, 72), g64(1, 72);   // N tiles of 256
  const int EB = 36864;  // 2*512*9216 / 256

  // b1/b2/b3 = BN(conv3x3(x, dil=1/2/4))
  gemm3x3_kernel<<<g512, 128, 0, stream>>>(W3, x, b3x3, B1f, 512, 512, 1, 0);
  bn_stats_kernel<<<512, 256, 0, stream>>>(B1f, STATS, 512);
  bn_apply_kernel<<<EB, 256, 0, stream>>>(B1f, STATS, bn0g, bn0b, B1f, 512, 0);
  gemm3x3_kernel<<<g512, 128, 0, stream>>>(W3, x, nullptr, B2f, 512, 512, 2, 0);
  bn_stats_kernel<<<512, 256, 0, stream>>>(B2f, STATS, 512);
  bn_apply_kernel<<<EB, 256, 0, stream>>>(B2f, STATS, bn1g, bn1b, B2f, 512, 0);
  gemm3x3_kernel<<<g512, 128, 0, stream>>>(W3, x, nullptr, B3f, 512, 512, 4, 0);
  bn_stats_kernel<<<512, 256, 0, stream>>>(B3f, STATS, 512);
  bn_apply_kernel<<<EB, 256, 0, stream>>>(B3f, STATS, bn2g, bn2b, B3f, 512, 0);

  // feat = relu(conv1x1(cat[b1,b2])); feat = relu(c31(feat)); att = softmax(c32)
  gemm1x1_kernel<<<g512, 128, 0, stream>>>(W1, B1f, B2f, b1x1, F512, 512, 1024, 512, 1);
  gemm3x3_kernel<<<g256, 128, 0, stream>>>(W31, F512, c31b, F256, 256, 512, 1, 1);
  c32_softmax_kernel<<<72, 256, 0, stream>>>(F256, c32w, c32b, ATT);

  // fusion = criss_cross(att0*b1, att1*b2)
  mul_att_kernel<<<EB, 256, 0, stream>>>(ATT, 0, B1f, XCC);
  mul_att_kernel<<<EB, 256, 0, stream>>>(ATT, 1, B2f, YCC);
  gemm1x1_kernel<<<g64, 128, 0, stream>>>(WQ, XCC, nullptr, qb, Qb, 64, 512, 512, 0);
  gemm1x1_kernel<<<g64, 128, 0, stream>>>(WK, XCC, nullptr, kb, Kb, 64, 512, 512, 0);
  gemm1x1_kernel<<<g512, 128, 0, stream>>>(WV, YCC, nullptr, vb, Vf, 512, 512, 512, 0);
  cc_attn_kernel<<<NPOS, 192, 0, stream>>>(Qb, Kb, A192);
  cc_out_kernel<<<EB, 256, 0, stream>>>(Vf, A192, XCC, ccg, FUS, 0);

  // feat1 path on cat[fusion,b3]
  gemm1x1_kernel<<<g512, 128, 0, stream>>>(W1, FUS, B3f, b1x1, F512, 512, 1024, 512, 1);
  gemm3x3_kernel<<<g256, 128, 0, stream>>>(W31, F512, c31b, F256, 256, 512, 1, 1);
  c32_softmax_kernel<<<72, 256, 0, stream>>>(F256, c32w, c32b, ATT);

  // ax = relu(criss_cross(sg*(att0*fusion+att1*b3), (1-sg)*x))
  fuse_sap_kernel<<<EB, 256, 0, stream>>>(ATT, FUS, B3f, x, sapg, XCC, YCC);
  gemm1x1_kernel<<<g64, 128, 0, stream>>>(WQ, XCC, nullptr, qb, Qb, 64, 512, 512, 0);
  gemm1x1_kernel<<<g64, 128, 0, stream>>>(WK, XCC, nullptr, kb, Kb, 64, 512, 512, 0);
  gemm1x1_kernel<<<g512, 128, 0, stream>>>(WV, YCC, nullptr, vb, Vf, 512, 512, 512, 0);
  cc_attn_kernel<<<NPOS, 192, 0, stream>>>(Qb, Kb, A192);
  cc_out_kernel<<<EB, 256, 0, stream>>>(Vf, A192, XCC, ccg, B1f, 1);

  // out = relu(BN(conv1x1(ax, last_w)))
  gemm1x1_kernel<<<g512, 128, 0, stream>>>(WL, B1f, nullptr, nullptr, B2f, 512, 512, 512, 0);
  bn_stats_kernel<<<512, 256, 0, stream>>>(B2f, STATS, 512);
  bn_apply_kernel<<<EB, 256, 0, stream>>>(B2f, STATS, lastg, lastb, (float*)d_out, 512, 1);
}
